// ShiftedWindowAttention_63823214018868
// MI455X (gfx1250) — matrix operand order repaired and checked by reading, not compiled, not run on hardware
//
#include <hip/hip_runtime.h>

// ---- CDNA5 WMMA types ----
typedef _Float16 v8h  __attribute__((ext_vector_type(8)));
typedef _Float16 v16h __attribute__((ext_vector_type(16)));
typedef float    v8f  __attribute__((ext_vector_type(8)));

#define NH   6
#define HD   32
#define NTOK 49
#define CCH  192
#define HH   112
#define WWD  112

// LDS pitches (in halves); all multiples of 8 -> 16B-aligned rows
#define XS_P 200   // x window / y buffer: 64 x 200
#define QK_P 40    // q,k per head: 64 x 40
#define VT_P 72    // v transposed per head: 32 x 72
#define P_P  72    // per-wave probability scratch: 16 x 72

// A-fragment (16x32 f16, M x K): lane l (m=l&15, half=l>>4):
//   halves [0..7]  = row[kk + half*8 .. +7]      (VGPR0..3 = K pairs)
//   halves [8..15] = row[kk + 16 + half*8 .. +7] (VGPR4..7)
__device__ __forceinline__ v16h frag_a(const _Float16* row, int kk, int half) {
  v8h lo = *(const v8h*)(row + kk + half * 8);
  v8h hi = *(const v8h*)(row + kk + 16 + half * 8);
  v16h r;
#pragma unroll
  for (int t = 0; t < 8; ++t) { r[t] = lo[t]; r[8 + t] = hi[t]; }
  return r;
}

// B-fragment (32x16 f16, K x N) from B^T row-major storage BT[n][k]:
//   lane l (n=l&15, half=l>>4): halves[0..7] = BT[n][kk + half*8 + j],
//   halves[8..15] = BT[n][kk + 16 + half*8 + j], j=0..7 (matches frag_a)
__device__ __forceinline__ v16h frag_b(const _Float16* rowT, int kk, int half) {
  v8h lo = *(const v8h*)(rowT + kk + half * 8);
  v8h hi = *(const v8h*)(rowT + kk + 16 + half * 8);
  v16h r;
#pragma unroll
  for (int t = 0; t < 8; ++t) { r[t] = lo[t]; r[8 + t] = hi[t]; }
  return r;
}

__device__ __forceinline__ v8f wmma_f16(v16h a, v16h b, v8f c) {
  return __builtin_amdgcn_wmma_f32_16x16x32_f16(false, a, false, b, (short)0, c,
                                                false, false);
}

// shift-mask region along one axis: only the last window (idx 15) mixes regions
__device__ __forceinline__ int region7(int widx, int i) {
  return (widx < 15) ? 0 : ((i < 4) ? 1 : 2);
}

__global__ void convert_weights_kernel(const float* __restrict__ wqkv,
                                       const float* __restrict__ wproj,
                                       _Float16* __restrict__ ws) {
  int i = blockIdx.x * 256 + threadIdx.x;
  if (i < 576 * CCH) ws[i] = (_Float16)wqkv[i];
  if (i < CCH * CCH) ws[576 * CCH + i] = (_Float16)wproj[i];
}

__global__ __launch_bounds__(256) void swin_attn_fused_kernel(
    const float* __restrict__ x,
    const _Float16* __restrict__ wqkv,   // [576][192] f16 (== B^T layout)
    const float* __restrict__ bqkv,      // [576]
    const _Float16* __restrict__ wproj,  // [192][192] f16 (== B^T layout)
    const float* __restrict__ bproj,     // [192]
    const float* __restrict__ relb,      // [169][6]
    float* __restrict__ out) {
  __shared__ _Float16 s_xs[64 * XS_P];        // x window (f16); later y = attn out
  __shared__ _Float16 s_q[NH * 64 * QK_P];    // q per head, row-major [m][d]
  __shared__ _Float16 s_k[NH * 64 * QK_P];    // k per head, row-major [m][d]
  __shared__ _Float16 s_vT[NH * HD * VT_P];   // v transposed [d][m]
  __shared__ _Float16 s_p[8 * 16 * P_P];      // per-wave softmax probs

  const int blk = blockIdx.x;
  const int b   = blk >> 8;
  const int w   = blk & 255;
  const int wy  = w >> 4;
  const int wx  = w & 15;
  const int tid  = threadIdx.x;
  const int lane = tid & 31;
  const int half = lane >> 4;
  const int l16  = lane & 15;
  // wave id as a guaranteed-scalar value -> scalar branches, no EXEC juggling
  const int wv = __builtin_amdgcn_readfirstlane(tid >> 5);
  const int rbw = wv & 3;        // this wave's fixed row-block
  const int cb0 = wv >> 2;       // starting col-block parity (0 or 1)

  // ---- Stage 1: gather shifted window, convert to f16, zero-pad rows 49..63
  const float* xb = x + (size_t)b * HH * WWD * CCH;
  for (int idx = tid; idx < NTOK * 48; idx += 256) {
    int row = idx / 48, c4 = idx % 48;
    int i = row / 7, j = row - i * 7;
    int rg = wy * 7 + i + 3; if (rg >= HH)  rg -= HH;
    int cg = wx * 7 + j + 3; if (cg >= WWD) cg -= WWD;
    float4 v = *(const float4*)(xb + ((size_t)rg * WWD + cg) * CCH + c4 * 4);
    _Float16* d = s_xs + row * XS_P + c4 * 4;
    d[0] = (_Float16)v.x; d[1] = (_Float16)v.y;
    d[2] = (_Float16)v.z; d[3] = (_Float16)v.w;
  }
  for (int idx = tid; idx < 15 * XS_P; idx += 256)
    s_xs[(NTOK + idx / XS_P) * XS_P + (idx % XS_P)] = (_Float16)0.f;
  __syncthreads();

  // ---- Stage 2: qkv = x @ Wqkv^T + b
  // wave handles row-block rbw for col-blocks cb0, cb0+2, ... (18 tiles)
  const float qscale = 0.17677669529663687f;  // 1/sqrt(32)
  {
    const _Float16* arow = s_xs + (rbw * 16 + l16) * XS_P;
    v16h afr[6];
#pragma unroll
    for (int k6 = 0; k6 < 6; ++k6) afr[k6] = frag_a(arow, k6 * 32, half);

    for (int cb = cb0; cb < 36; cb += 2) {
      const _Float16* brow = wqkv + (size_t)(cb * 16 + l16) * CCH;
      v8f acc = {0.f, 0.f, 0.f, 0.f, 0.f, 0.f, 0.f, 0.f};
#pragma unroll
      for (int k6 = 0; k6 < 6; ++k6)
        acc = wmma_f16(afr[k6], frag_b(brow, k6 * 32, half), acc);

      float bias = bqkv[cb * 16 + l16];
      int cbm = cb % 12;               // scalar: position within segment
      int h   = cbm >> 1;              // head (uniform per tile)
      int d   = (cb & 1) * 16 + l16;   // channel within head
      int m0  = rbw * 16 + half * 8;   // first m for this lane's 8 elements
      if (cb < 12) {                   // ---- q (scaled)
        _Float16* dst = s_q + ((h * 64 + m0) * QK_P + d);
#pragma unroll
        for (int r = 0; r < 8; ++r)
          dst[r * QK_P] = (_Float16)((acc[r] + bias) * qscale);
      } else if (cb < 24) {            // ---- k
        _Float16* dst = s_k + ((h * 64 + m0) * QK_P + d);
#pragma unroll
        for (int r = 0; r < 8; ++r)
          dst[r * QK_P] = (_Float16)(acc[r] + bias);
      } else {                         // ---- v, stored transposed: contiguous m
        v8h pk;
#pragma unroll
        for (int r = 0; r < 8; ++r) pk[r] = (_Float16)(acc[r] + bias);
        *(v8h*)(s_vT + (h * HD + d) * VT_P + m0) = pk;   // one ds_store_b128
      }
    }
  }
  __syncthreads();

  // ---- Stage 3: per (head, row-block) strip: QK^T + bias + mask, softmax, PV
  for (int s = wv; s < 24; s += 8) {
    int h  = s >> 2;       // scalar
    int rb = s & 3;        // scalar (== rbw)
    const _Float16* qrow = s_q + (h * 64 + rb * 16 + l16) * QK_P;
    v16h aq = frag_a(qrow, 0, half);
    v8f S[4];
#pragma unroll
    for (int cb = 0; cb < 4; ++cb) {
      const _Float16* krow = s_k + (h * 64 + cb * 16 + l16) * QK_P;
      v8f z = {0.f, 0.f, 0.f, 0.f, 0.f, 0.f, 0.f, 0.f};
      S[cb] = wmma_f16(aq, frag_b(krow, 0, half), z);
    }
    // bias + shift-mask + invalid-column masking
#pragma unroll
    for (int cb = 0; cb < 4; ++cb) {
      int mm = cb * 16 + l16;
      int i2 = 0, j2 = 0, cnt2 = 0;
      bool colv = (mm < NTOK);
      if (colv) {
        i2 = mm / 7; j2 = mm - i2 * 7;
        cnt2 = region7(wy, i2) * 3 + region7(wx, j2);
      }
#pragma unroll
      for (int r = 0; r < 8; ++r) {
        int m = rb * 16 + r + half * 8;
        float v = S[cb][r];
        if (!colv) {
          v = -1e30f;
        } else if (m < NTOK) {
          int i1 = m / 7, j1 = m - i1 * 7;
          int idx = (i1 - i2 + 6) * 13 + (j1 - j2 + 6);
          v += relb[idx * NH + h];
          int cnt1 = region7(wy, i1) * 3 + region7(wx, j1);
          if (cnt1 != cnt2) v -= 100.f;
        }
        S[cb][r] = v;
      }
    }
    // softmax per row (rows live per-VGPR, cols across a 16-lane half)
#pragma unroll
    for (int r = 0; r < 8; ++r) {
      float mx = fmaxf(fmaxf(S[0][r], S[1][r]), fmaxf(S[2][r], S[3][r]));
      mx = fmaxf(mx, __shfl_xor(mx, 1));
      mx = fmaxf(mx, __shfl_xor(mx, 2));
      mx = fmaxf(mx, __shfl_xor(mx, 4));
      mx = fmaxf(mx, __shfl_xor(mx, 8));
      float sum = 0.f;
#pragma unroll
      for (int cb = 0; cb < 4; ++cb) {
        float e = __expf(S[cb][r] - mx);
        S[cb][r] = e; sum += e;
      }
      sum += __shfl_xor(sum, 1); sum += __shfl_xor(sum, 2);
      sum += __shfl_xor(sum, 4); sum += __shfl_xor(sum, 8);
      float inv = 1.f / sum;
#pragma unroll
      for (int cb = 0; cb < 4; ++cb)
        s_p[(wv * 16 + r + half * 8) * P_P + cb * 16 + l16] =
            (_Float16)(S[cb][r] * inv);
    }
    // O = P @ v  (A from per-wave scratch; same-wave LDS ops are in order)
    const _Float16* prow = s_p + (wv * 16 + l16) * P_P;
    v16h a0 = frag_a(prow, 0, half);
    v16h a1 = frag_a(prow, 32, half);
#pragma unroll
    for (int cb2 = 0; cb2 < 2; ++cb2) {
      const _Float16* vrow = s_vT + (h * HD + cb2 * 16 + l16) * VT_P;
      v8f O = {0.f, 0.f, 0.f, 0.f, 0.f, 0.f, 0.f, 0.f};
      O = wmma_f16(a0, frag_b(vrow, 0, half), O);
      O = wmma_f16(a1, frag_b(vrow, 32, half), O);
#pragma unroll
      for (int r = 0; r < 8; ++r) {
        int m = rb * 16 + r + half * 8;
        s_xs[m * XS_P + h * HD + cb2 * 16 + l16] = (_Float16)O[r];
      }
    }
  }
  __syncthreads();

  // ---- Stage 4: out = y @ Wproj^T + b, scatter to rolled-back positions
  {
    const _Float16* arow = s_xs + (rbw * 16 + l16) * XS_P;
    v16h afr[6];
#pragma unroll
    for (int k6 = 0; k6 < 6; ++k6) afr[k6] = frag_a(arow, k6 * 32, half);

    for (int cb = cb0; cb < 12; cb += 2) {
      const _Float16* brow = wproj + (size_t)(cb * 16 + l16) * CCH;
      v8f acc = {0.f, 0.f, 0.f, 0.f, 0.f, 0.f, 0.f, 0.f};
#pragma unroll
      for (int k6 = 0; k6 < 6; ++k6)
        acc = wmma_f16(afr[k6], frag_b(brow, k6 * 32, half), acc);

      int o = cb * 16 + l16;
      float bias = bproj[o];
#pragma unroll
      for (int r = 0; r < 8; ++r) {
        int m = rbw * 16 + r + half * 8;
        if (m < NTOK) {
          int i = m / 7, j = m - i * 7;
          int rg = wy * 7 + i + 3; if (rg >= HH)  rg -= HH;
          int cg = wx * 7 + j + 3; if (cg >= WWD) cg -= WWD;
          out[(((size_t)b * HH + rg) * WWD + cg) * CCH + o] = acc[r] + bias;
        }
      }
    }
  }
}

extern "C" void kernel_launch(void* const* d_in, const int* in_sizes, int n_in,
                              void* d_out, int out_size, void* d_ws, size_t ws_size,
                              hipStream_t stream) {
  (void)in_sizes; (void)n_in; (void)out_size; (void)ws_size;
  const float* x     = (const float*)d_in[0];
  const float* wqkv  = (const float*)d_in[1];
  const float* bqkv  = (const float*)d_in[2];
  const float* wproj = (const float*)d_in[3];
  const float* bproj = (const float*)d_in[4];
  const float* relb  = (const float*)d_in[5];
  _Float16* wsh = (_Float16*)d_ws;   // [576*192] qkv_f16 then [192*192] proj_f16

  convert_weights_kernel<<<(576 * CCH + 255) / 256, 256, 0, stream>>>(
      wqkv, wproj, wsh);
  swin_attn_fused_kernel<<<16 * 256, 256, 0, stream>>>(
      x, wsh, bqkv, wsh + 576 * CCH, bproj, relb, (float*)d_out);
}
